// MSABlock_53867479826843
// MI455X (gfx1250) — matrix operand order repaired and checked by reading, not compiled, not run on hardware
//
#include <hip/hip_runtime.h>
#include <hip/hip_bf16.h>

// ---------------- dims ----------------
#define S_  512
#define N_  256
#define CM_ 64
#define CZ_ 128
#define CH_ 32
#define H_  8
#define DH_ 32
#define FF_ 256
#define SC_ 128   // s-chunk for PWA value/attention buffers (4 chunks)

// padded LDS strides (elements); multiples of 16 keep b128 alignment,
// odd multiple of 16 banks avoids conflicts on transposed reads
#define XLD_ 1040   // ldsX row stride (1024 + 16)
#define VLD_ 272    // ldsVt row stride (256 + 16)

typedef __attribute__((ext_vector_type(16))) __bf16 v16bf;
typedef __attribute__((ext_vector_type(8)))  float  v8f;

union AFrag { v16bf v; uint4 q[2]; };

__device__ inline __bf16 f2bf(float f) {
  unsigned u = __builtin_bit_cast(unsigned, f);
  u += 0x7FFFu + ((u >> 16) & 1u);
  unsigned short s = (unsigned short)(u >> 16);
  return __builtin_bit_cast(__bf16, s);
}
__device__ inline __bf16 u16bf(unsigned short s) { return __builtin_bit_cast(__bf16, s); }
__device__ inline float bf2f(__bf16 h) {
  unsigned short s = __builtin_bit_cast(unsigned short, h);
  unsigned u = ((unsigned)s) << 16;
  return __builtin_bit_cast(float, u);
}
__device__ inline v8f zero8() { v8f r; for (int i = 0; i < 8; ++i) r[i] = 0.0f; return r; }

__device__ inline v8f wmma_bf16(v16bf a, v16bf b, v8f c) {
  return __builtin_amdgcn_wmma_f32_16x16x32_bf16(false, a, false, b, (short)0, c, false, false);
}

__device__ inline float wave_sum(float v) {
  for (int off = 16; off; off >>= 1) v += __shfl_xor(v, off, 32);
  return v;
}
__device__ inline float wave_max(float v) {
  for (int off = 16; off; off >>= 1) v = fmaxf(v, __shfl_xor(v, off, 32));
  return v;
}

// ---------------- wout (1024x128) -> transposed bf16 woutT[o][k] ----------------
__global__ void k_cvt_woutT(const float* __restrict__ in, __bf16* __restrict__ out, int n) {
  int t = blockIdx.x * 256 + threadIdx.x;
  if (t < n) {
    int k = t >> 7, o = t & 127;            // in is [k][o], k<1024, o<128
    out[(size_t)o * 1024 + k] = f2bf(in[t]);
  }
}

// ---------------- OPM prep: LN(m); aT[(i*32+c)][s], bT[(j*32+e)][s] (both K-major bf16) ----------------
__global__ void k_opm_prep(const float* __restrict__ m, const float* __restrict__ mask,
                           const float* __restrict__ lnw, const float* __restrict__ lnb,
                           const float* __restrict__ w1, const float* __restrict__ w2,
                           __bf16* __restrict__ aT, __bf16* __restrict__ bT) {
  __shared__ float smn[8][CM_];
  int wave = threadIdx.x >> 5, lane = threadIdx.x & 31;
  int row = blockIdx.x * 8 + wave;       // [0, S*N)
  int s = row >> 8, n = row & 255;
  const float* mr = m + (size_t)row * CM_;
  float x0 = mr[lane], x1 = mr[lane + 32];
  float mu = wave_sum(x0 + x1) * (1.0f / 64.0f);
  float d0 = x0 - mu, d1 = x1 - mu;
  float var = wave_sum(d0 * d0 + d1 * d1) * (1.0f / 64.0f);
  float rstd = rsqrtf(var + 1e-5f);
  smn[wave][lane]      = d0 * rstd * lnw[lane]      + lnb[lane];
  smn[wave][lane + 32] = d1 * rstd * lnw[lane + 32] + lnb[lane + 32];
  __syncthreads();
  float mk = mask[s * N_ + n];
  float acc1 = 0.f, acc2 = 0.f;
  for (int k = 0; k < CM_; ++k) {
    float v = smn[wave][k];
    acc1 += v * w1[k * CH_ + lane];
    acc2 += v * w2[k * CH_ + lane];
  }
  acc1 *= mk; acc2 *= mk;
  aT[((size_t)n * CH_ + lane) * S_ + s] = f2bf(acc1);
  bT[((size_t)n * CH_ + lane) * S_ + s] = f2bf(acc2);
}

// ---------------- norm[i][j] = sum_s mask[s,i]*mask[s,j] ----------------
__global__ void k_norm(const float* __restrict__ mask, float* __restrict__ norm) {
  int t = blockIdx.x * 256 + threadIdx.x;   // N*N
  int i = t >> 8, j = t & 255;
  float acc = 0.f;
  for (int s = 0; s < S_; ++s) acc += mask[s * N_ + i] * mask[s * N_ + j];
  norm[t] = acc;
}

// ---------------- fused OPM GEMMs + z epilogue ----------------
// grid: (jb=16, i=256), block 256 (8 waves)
__global__ void __launch_bounds__(256) k_opm_gemm(
    const __bf16* __restrict__ aT, const __bf16* __restrict__ bT,
    const __bf16* __restrict__ woutT, const float* __restrict__ bout,
    const float* __restrict__ norm, const float* __restrict__ zin,
    float* __restrict__ zout) {
  __shared__ __bf16 ldsX[16 * XLD_];   // transposed outer tile X[jj][c*32+e] (padded)
  const int jb = blockIdx.x, i = blockIdx.y;
  const int tid = threadIdx.x, wave = tid >> 5, lane = tid & 31;
  const int lh = lane & 15, hi = lane >> 4;
  const int c0 = (wave & 1) * 16;      // M tile (c rows)
  const int ng = wave >> 1;            // group of 8 N tiles

  v8f acc[8];
  for (int t = 0; t < 8; ++t) acc[t] = zero8();

  const __bf16* arow = aT + ((size_t)i * CH_ + c0 + lh) * S_;
  // this wave's 8 B columns (K-major rows of bT)
  const __bf16* bcol[8];
  for (int t = 0; t < 8; ++t)
    bcol[t] = bT + ((size_t)jb * 512 + (ng * 8 + t) * 16 + lh) * S_;

  for (int ks = 0; ks < 16; ++ks) {    // K = 512, step 32
    const int k0 = ks * 32;
    AFrag af;
    {
      const __bf16* ap = arow + k0 + hi * 16;
      __builtin_prefetch(arow + k0 + 32, 0, 1);
      af.q[0] = *(const uint4*)(ap);
      af.q[1] = *(const uint4*)(ap + 8);
    }
    for (int t = 0; t < 8; ++t) {
      AFrag bf_;
      const __bf16* bp = bcol[t] + k0 + hi * 16;   // 16 contiguous K values
      bf_.q[0] = *(const uint4*)(bp);
      bf_.q[1] = *(const uint4*)(bp + 8);
      acc[t] = wmma_bf16(af.v, bf_.v, acc[t]);
    }
  }
  // scatter D tiles into ldsX as X[jj][c*32+e] (bf16, padded rows)
  for (int t = 0; t < 8; ++t) {
    const int ncol = (ng * 8 + t) * 16 + lh;  // local col 0..511 = jj*32+e
    const int jj = ncol >> 5, e = ncol & 31;
    const int mbase = c0 + hi * 8;
    for (int g = 0; g < 8; ++g)
      ldsX[jj * XLD_ + (mbase + g) * 32 + e] = f2bf(acc[t][g]);
  }
  __syncthreads();
  // GEMM2: Z[16 x 128] = X(16x1024) @ wout(1024x128); wave -> one 16-col tile of o
  const int n0 = wave * 16;
  v8f acc2 = zero8();
  const __bf16* wcol = woutT + (size_t)(n0 + lh) * 1024;  // K-major wout column
  for (int ks = 0; ks < 32; ++ks) {
    const int k0 = ks * 32;
    AFrag af, bf_;
    const __bf16* xp = ldsX + lh * XLD_ + k0 + hi * 16;
    af.q[0] = *(const uint4*)xp;
    af.q[1] = *(const uint4*)(xp + 8);
    const __bf16* bp = wcol + k0 + hi * 16;
    bf_.q[0] = *(const uint4*)(bp);
    bf_.q[1] = *(const uint4*)(bp + 8);
    acc2 = wmma_bf16(af.v, bf_.v, acc2);
  }
  const int o = n0 + lh;
  const float bo = bout[o];
  for (int g = 0; g < 8; ++g) {
    const int j = jb * 16 + g + hi * 8;
    const float nrm = norm[i * N_ + j];
    const size_t idx = ((size_t)i * N_ + j) * CZ_ + o;
    zout[idx] = zin[idx] + (acc2[g] + bo) / (nrm + 1e-3f);
  }
}

// ---------------- PWA logits: LN(z_new) @ wz ----------------
__global__ void k_logits(const float* __restrict__ z, const float* __restrict__ lnw,
                         const float* __restrict__ lnb, const float* __restrict__ wz,
                         float* __restrict__ logits) {
  __shared__ float szn[8][CZ_];
  int wave = threadIdx.x >> 5, lane = threadIdx.x & 31;
  int row = blockIdx.x * 8 + wave;     // [0, N*N)
  const float* zr = z + (size_t)row * CZ_;
  float x[4]; float sum = 0.f;
  for (int q = 0; q < 4; ++q) { x[q] = zr[lane + 32 * q]; sum += x[q]; }
  float mu = wave_sum(sum) * (1.0f / 128.0f);
  float vs = 0.f;
  for (int q = 0; q < 4; ++q) { x[q] -= mu; vs += x[q] * x[q]; }
  float rstd = rsqrtf(wave_sum(vs) * (1.0f / 128.0f) + 1e-5f);
  for (int q = 0; q < 4; ++q)
    szn[wave][lane + 32 * q] = x[q] * rstd * lnw[lane + 32 * q] + lnb[lane + 32 * q];
  __syncthreads();
  if (lane < H_) {
    float acc = 0.f;
    for (int k = 0; k < CZ_; ++k) acc += szn[wave][k] * wz[k * H_ + lane];
    logits[(size_t)row * H_ + lane] = acc;
  }
}

// ---------------- softmax over j -> watt[h][i][j] (bf16) ----------------
__global__ void k_softmax(const float* __restrict__ logits, __bf16* __restrict__ watt) {
  int wave = threadIdx.x >> 5, lane = threadIdx.x & 31;
  int r = blockIdx.x * 8 + wave;       // [0, N*H) : r = i*8+h
  int i = r >> 3, h = r & 7;
  float v[8]; float mx = -1e30f;
  for (int q = 0; q < 8; ++q) {
    int j = lane + 32 * q;
    v[q] = logits[((size_t)i * N_ + j) * H_ + h];
    mx = fmaxf(mx, v[q]);
  }
  mx = wave_max(mx);
  float sum = 0.f;
  for (int q = 0; q < 8; ++q) { v[q] = __expf(v[q] - mx); sum += v[q]; }
  sum = wave_sum(sum);
  float inv = 1.0f / sum;
  for (int q = 0; q < 8; ++q) {
    int j = lane + 32 * q;
    watt[((size_t)h * N_ + i) * N_ + j] = f2bf(v[q] * inv);
  }
}

// ---------------- PWA v/g projections for an s-chunk ----------------
__global__ void k_vg(const float* __restrict__ m, const float* __restrict__ lnw,
                     const float* __restrict__ lnb, const float* __restrict__ wv,
                     const float* __restrict__ wg, __bf16* __restrict__ vout,
                     __bf16* __restrict__ gout, int s0) {
  __shared__ float smn[8][CM_];
  int wave = threadIdx.x >> 5, lane = threadIdx.x & 31;
  int row = blockIdx.x * 8 + wave;     // [0, SC*N)
  int sl = row >> 8, n = row & 255;
  const float* mr = m + (((size_t)(s0 + sl)) * N_ + n) * CM_;
  float x0 = mr[lane], x1 = mr[lane + 32];
  float mu = wave_sum(x0 + x1) * (1.0f / 64.0f);
  float d0 = x0 - mu, d1 = x1 - mu;
  float rstd = rsqrtf(wave_sum(d0 * d0 + d1 * d1) * (1.0f / 64.0f) + 1e-5f);
  smn[wave][lane]      = d0 * rstd * lnw[lane]      + lnb[lane];
  smn[wave][lane + 32] = d1 * rstd * lnw[lane + 32] + lnb[lane + 32];
  __syncthreads();
  for (int q = 0; q < H_; ++q) {
    const int c = lane + 32 * q;       // h = q, d = lane
    float av = 0.f, ag = 0.f;
    for (int k = 0; k < CM_; ++k) {
      float x = smn[wave][k];
      av += x * wv[k * (H_ * DH_) + c];
      ag += x * wg[k * (H_ * DH_) + c];
    }
    float g = 1.0f / (1.0f + __expf(-ag));
    size_t idx = (((size_t)(sl * H_ + q)) * N_ + n) * DH_ + lane;
    vout[idx] = f2bf(av);
    gout[idx] = f2bf(g);
  }
}

// ---------------- WMMA attention: o[i,d] = sum_j watt[h][i][j] * v[j,d] per (s,h) ----------------
__device__ inline void store_tile(__bf16* o, size_t base, int m0, int n0, int lh, int hi, const v8f& a) {
  const int d = n0 + lh;
  for (int g = 0; g < 8; ++g) {
    const int ii = m0 + g + hi * 8;
    o[base + (size_t)ii * DH_ + d] = f2bf(a[g]);
  }
}

__global__ void __launch_bounds__(256) k_att(const __bf16* __restrict__ watt,
                                             const __bf16* __restrict__ v,
                                             __bf16* __restrict__ o) {
  __shared__ __bf16 ldsVt[DH_ * VLD_];  // V transposed: [d][j], padded rows
  const int b = blockIdx.x;             // [0, SC*H)
  const int sl = b >> 3, h = b & 7;
  const int tid = threadIdx.x, wave = tid >> 5, lane = tid & 31;
  const int lh = lane & 15, hi = lane >> 4;
  const size_t vbase = ((size_t)(sl * H_ + h)) * N_ * DH_;
  { // stage V (256x32 bf16), transposing to [d][j] in LDS
    const unsigned* src = (const unsigned*)(v + vbase);
    for (int r = 0; r < 16; ++r) {
      int dw = tid + 256 * r;           // 0..4095 dwords of the row-major tile
      int j  = dw >> 4;                 // 16 dwords per j-row (32 d)
      int d0 = (dw & 15) * 2;
      unsigned val = src[dw];
      ldsVt[d0 * VLD_ + j]       = u16bf((unsigned short)(val & 0xFFFFu));
      ldsVt[(d0 + 1) * VLD_ + j] = u16bf((unsigned short)(val >> 16));
    }
  }
  __syncthreads();
  v8f a00 = zero8(), a01 = zero8(), a10 = zero8(), a11 = zero8();
  const __bf16* wbase = watt + ((size_t)h * N_) * N_;
  for (int ks = 0; ks < 8; ++ks) {     // K = 256
    const int k0 = ks * 32;
    AFrag b0, b1;
    {
      const __bf16* p0 = ldsVt + lh * VLD_ + k0 + hi * 16;
      b0.q[0] = *(const uint4*)p0;
      b0.q[1] = *(const uint4*)(p0 + 8);
      const __bf16* p1 = ldsVt + (16 + lh) * VLD_ + k0 + hi * 16;
      b1.q[0] = *(const uint4*)p1;
      b1.q[1] = *(const uint4*)(p1 + 8);
    }
    AFrag af;
    const __bf16* ap = wbase + (size_t)(wave * 16 + lh) * N_ + k0 + hi * 16;
    af.q[0] = *(const uint4*)ap; af.q[1] = *(const uint4*)(ap + 8);
    a00 = wmma_bf16(af.v, b0.v, a00);
    a01 = wmma_bf16(af.v, b1.v, a01);
    const __bf16* ap2 = wbase + (size_t)((wave + 8) * 16 + lh) * N_ + k0 + hi * 16;
    af.q[0] = *(const uint4*)ap2; af.q[1] = *(const uint4*)(ap2 + 8);
    a10 = wmma_bf16(af.v, b0.v, a10);
    a11 = wmma_bf16(af.v, b1.v, a11);
  }
  store_tile(o, vbase, wave * 16, 0, lh, hi, a00);
  store_tile(o, vbase, wave * 16, 16, lh, hi, a01);
  store_tile(o, vbase, (wave + 8) * 16, 0, lh, hi, a10);
  store_tile(o, vbase, (wave + 8) * 16, 16, lh, hi, a11);
}

// ---------------- PWA output: m_out = m + (g*o) @ wo ----------------
__global__ void k_pwa_out(const float* __restrict__ min_, const __bf16* __restrict__ g,
                          const __bf16* __restrict__ o, const float* __restrict__ wo,
                          float* __restrict__ mout, int s0) {
  __shared__ float gt[8][H_ * DH_];
  int wave = threadIdx.x >> 5, lane = threadIdx.x & 31;
  int row = blockIdx.x * 8 + wave;     // [0, SC*N)
  int sl = row >> 8, i = row & 255;
  for (int q = 0; q < H_; ++q) {
    size_t idx = (((size_t)(sl * H_ + q)) * N_ + i) * DH_ + lane;
    gt[wave][q * 32 + lane] = bf2f(g[idx]) * bf2f(o[idx]);
  }
  __syncthreads();
  for (int half = 0; half < 2; ++half) {
    const int oc = lane + 32 * half;
    float acc = 0.f;
    for (int k = 0; k < H_ * DH_; ++k) acc += gt[wave][k] * wo[k * CM_ + oc];
    size_t gi = (((size_t)(s0 + sl)) * N_ + i) * CM_ + oc;
    mout[gi] = min_[gi] + acc;
  }
}

// ---------------- Transition (SwiGLU), in-place on m_out ----------------
__global__ void k_transition(float* __restrict__ mio, const float* __restrict__ lnw,
                             const float* __restrict__ lnb, const float* __restrict__ wa,
                             const float* __restrict__ wb, const float* __restrict__ wo) {
  __shared__ float sx[8][CM_];
  __shared__ float st[8][FF_];
  int wave = threadIdx.x >> 5, lane = threadIdx.x & 31;
  size_t row = (size_t)blockIdx.x * 8 + wave;
  float* mr = mio + row * CM_;
  float x0 = mr[lane], x1 = mr[lane + 32];
  float mu = wave_sum(x0 + x1) * (1.0f / 64.0f);
  float d0 = x0 - mu, d1 = x1 - mu;
  float rstd = rsqrtf(wave_sum(d0 * d0 + d1 * d1) * (1.0f / 64.0f) + 1e-5f);
  sx[wave][lane]      = d0 * rstd * lnw[lane]      + lnb[lane];
  sx[wave][lane + 32] = d1 * rstd * lnw[lane + 32] + lnb[lane + 32];
  __syncthreads();
  for (int q = 0; q < 8; ++q) {
    const int f = lane + 32 * q;
    float aa = 0.f, bb = 0.f;
    for (int k = 0; k < CM_; ++k) {
      float xv = sx[wave][k];
      aa += xv * wa[k * FF_ + f];
      bb += xv * wb[k * FF_ + f];
    }
    st[wave][f] = (aa / (1.0f + __expf(-aa))) * bb;
  }
  __syncthreads();
  for (int half = 0; half < 2; ++half) {
    const int oc = lane + 32 * half;
    float acc = 0.f;
    for (int k = 0; k < FF_; ++k) acc += st[wave][k] * wo[k * CM_ + oc];
    mr[oc] = (half ? x1 : x0) + acc;
  }
}

// ---------------- host ----------------
extern "C" void kernel_launch(void* const* d_in, const int* in_sizes, int n_in,
                              void* d_out, int out_size, void* d_ws, size_t ws_size,
                              hipStream_t stream) {
  const float* m        = (const float*)d_in[0];
  const float* z        = (const float*)d_in[1];
  const float* mask     = (const float*)d_in[2];
  const float* opm_ln_w = (const float*)d_in[3];
  const float* opm_ln_b = (const float*)d_in[4];
  const float* opm_w1   = (const float*)d_in[5];
  const float* opm_w2   = (const float*)d_in[6];
  const float* opm_wout = (const float*)d_in[7];
  const float* opm_bout = (const float*)d_in[8];
  const float* lnm_w    = (const float*)d_in[9];
  const float* lnm_b    = (const float*)d_in[10];
  const float* lnz_w    = (const float*)d_in[11];
  const float* lnz_b    = (const float*)d_in[12];
  const float* pwa_wv   = (const float*)d_in[13];
  const float* pwa_wz   = (const float*)d_in[14];
  const float* pwa_wg   = (const float*)d_in[15];
  const float* pwa_wo   = (const float*)d_in[16];
  const float* tr_ln_w  = (const float*)d_in[17];
  const float* tr_ln_b  = (const float*)d_in[18];
  const float* tr_wa    = (const float*)d_in[19];
  const float* tr_wb    = (const float*)d_in[20];
  const float* tr_wo    = (const float*)d_in[21];

  float* m_out = (float*)d_out;
  float* z_out = m_out + (size_t)S_ * N_ * CM_;

  char* ws = (char*)d_ws;
  const size_t AT_OFF     = 0;                          // 8 MB  bf16 aT[i*32+c][s]
  const size_t B_OFF      = AT_OFF + 8388608;           // 8 MB  bf16 bT[j*32+e][s]
  const size_t WOUT_OFF   = B_OFF + 8388608;            // 256 KB bf16 woutT[o][k]
  const size_t NORM_OFF   = WOUT_OFF + 262144;          // 256 KB f32 norm
  const size_t LOGITS_OFF = NORM_OFF + 262144;          // 2 MB  f32 logits
  const size_t WATT_OFF   = LOGITS_OFF + 2097152;       // 1 MB  bf16 watt
  const size_t V_OFF      = WATT_OFF + 1048576;         // 16 MB bf16 v chunk
  const size_t G_OFF      = V_OFF + 16777216;           // 16 MB bf16 g chunk
  const size_t O_OFF      = G_OFF + 16777216;           // 16 MB bf16 o chunk

  __bf16* aT      = (__bf16*)(ws + AT_OFF);
  __bf16* bT      = (__bf16*)(ws + B_OFF);
  __bf16* woutT   = (__bf16*)(ws + WOUT_OFF);
  float*  norm    = (float*)(ws + NORM_OFF);
  float*  logits  = (float*)(ws + LOGITS_OFF);
  __bf16* watt    = (__bf16*)(ws + WATT_OFF);
  __bf16* vbuf    = (__bf16*)(ws + V_OFF);
  __bf16* gbuf    = (__bf16*)(ws + G_OFF);
  __bf16* obuf    = (__bf16*)(ws + O_OFF);

  // ---- OuterProductMean ----
  k_cvt_woutT<<<(CH_ * CH_ * CZ_ + 255) / 256, 256, 0, stream>>>(opm_wout, woutT, CH_ * CH_ * CZ_);
  k_opm_prep<<<S_ * N_ / 8, 256, 0, stream>>>(m, mask, opm_ln_w, opm_ln_b, opm_w1, opm_w2, aT, bT);
  k_norm<<<N_ * N_ / 256, 256, 0, stream>>>(mask, norm);
  k_opm_gemm<<<dim3(16, 256), 256, 0, stream>>>(aT, bT, woutT, opm_bout, norm, z, z_out);

  // ---- PWA attention weights from updated z ----
  k_logits<<<N_ * N_ / 8, 256, 0, stream>>>(z_out, lnz_w, lnz_b, pwa_wz, logits);
  k_softmax<<<N_ * H_ / 8, 256, 0, stream>>>(logits, watt);

  // ---- PWA main, chunked over s ----
  for (int c = 0; c < S_ / SC_; ++c) {
    const int s0 = c * SC_;
    k_vg<<<SC_ * N_ / 8, 256, 0, stream>>>(m, lnm_w, lnm_b, pwa_wv, pwa_wg, vbuf, gbuf, s0);
    k_att<<<SC_ * H_, 256, 0, stream>>>(watt, vbuf, obuf);
    k_pwa_out<<<SC_ * N_ / 8, 256, 0, stream>>>(m, gbuf, obuf, pwa_wo, m_out, s0);
  }

  // ---- Transition ----
  k_transition<<<S_ * N_ / 8, 256, 0, stream>>>(m_out, tr_ln_w, tr_ln_b, tr_wa, tr_wb, tr_wo);
}
